// PosPoolSE_23527830847986
// MI455X (gfx1250) — matrix operand order repaired and checked by reading, not compiled, not run on hardware
//
#include <hip/hip_runtime.h>
#include <hip/hip_fp16.h>

#define BB 4
#define N1V 2048
#define N2V 8192
#define CCV 288
#define NSV 32
#define RADIUSV 0.2f
#define EPSV 1e-5f

typedef __attribute__((ext_vector_type(16))) _Float16 v16h;
typedef __attribute__((ext_vector_type(8)))  float    v8f;
typedef __attribute__((ext_vector_type(4)))  int      v4i;

#if defined(__has_builtin)
#if __has_builtin(__builtin_amdgcn_global_load_async_to_lds_b128)
#define HAVE_ASYNC_LDS 1
#endif
#endif

#if defined(HAVE_ASYNC_LDS)
static __device__ __forceinline__ void pp_wait_async0() {
#if __has_builtin(__builtin_amdgcn_s_wait_asynccnt)
  __builtin_amdgcn_s_wait_asynccnt(0);
#else
  asm volatile("s_wait_asynccnt 0x0" ::: "memory");
#endif
}
#endif

// ---------------- K0: zero accumulators (must run every call) ----------------
__global__ void pp_init(float* __restrict__ gse, float* __restrict__ chSum,
                        float* __restrict__ chSumSq) {
  int i = blockIdx.x * 256 + threadIdx.x;
  if (i < BB * CCV) gse[i] = 0.f;
  if (i < CCV) { chSum[i] = 0.f; chSumSq[i] = 0.f; }
}

// ---------------- K1: masked ball query (async-staged LDS chunks) ------------
__global__ __launch_bounds__(256) void pp_ballquery(
    const float* __restrict__ qxyz, const float* __restrict__ sxyz,
    const float* __restrict__ qmask, const float* __restrict__ smask,
    int* __restrict__ idxBuf, float* __restrict__ fmBuf, float* __restrict__ fmSum) {
  constexpr int CHUNK = 1024;
  __shared__ __align__(16) float sPts[CHUNK * 3];  // 12 KB, contiguous copy of sxyz chunk
  __shared__ __align__(16) float sMsk[CHUNK];      //  4 KB, contiguous copy of smask chunk
  const int gq = blockIdx.x * 256 + threadIdx.x;   // 0 .. B*N1-1
  const int b  = gq / N1V;
  const float qm = qmask[gq];
  const float qx = qxyz[(size_t)gq * 3 + 0];
  const float qy = qxyz[(size_t)gq * 3 + 1];
  const float qz = qxyz[(size_t)gq * 3 + 2];
  const float r2 = RADIUSV * RADIUSV;
  int cnt = 0, firstIdx = 0;
  int* myIdx = idxBuf + (size_t)gq * NSV;

  for (int c0 = 0; c0 < N2V; c0 += CHUNK) {
    __syncthreads();
#if defined(HAVE_ASYNC_LDS)
    {
      const float* gp = sxyz + ((size_t)b * N2V + c0) * 3;   // 16B-aligned chunk base
      for (int t = threadIdx.x; t < (CHUNK * 3) / 4; t += 256)
        __builtin_amdgcn_global_load_async_to_lds_b128(
            (v4i*)(gp + t * 4), (v4i*)(&sPts[t * 4]), 0, 0);
      const float* gm = smask + (size_t)b * N2V + c0;
      for (int t = threadIdx.x; t < CHUNK / 4; t += 256)
        __builtin_amdgcn_global_load_async_to_lds_b128(
            (v4i*)(gm + t * 4), (v4i*)(&sMsk[t * 4]), 0, 0);
      pp_wait_async0();
    }
#else
    for (int t = threadIdx.x; t < CHUNK * 3; t += 256)
      sPts[t] = sxyz[((size_t)b * N2V + c0) * 3 + t];
    for (int t = threadIdx.x; t < CHUNK; t += 256)
      sMsk[t] = smask[(size_t)b * N2V + c0 + t];
#endif
    __syncthreads();
    if (qm > 0.f && cnt < NSV) {
      for (int t = 0; t < CHUNK; ++t) {
        const float dx = sPts[t * 3 + 0] - qx;
        const float dy = sPts[t * 3 + 1] - qy;
        const float dz = sPts[t * 3 + 2] - qz;
        const float d2 = dx * dx + dy * dy + dz * dz;
        if (d2 < r2 && sMsk[t] > 0.f) {
          if (cnt == 0) firstIdx = c0 + t;
          myIdx[cnt] = c0 + t;
          if (++cnt == NSV) break;
        }
      }
    }
  }
  const float inv = 1.f - qm;   // qm==0 -> fm = 1 everywhere, idx = 0 (matches ref)
  for (int s = 0; s < NSV; ++s) {
    if (s >= cnt) myIdx[s] = firstIdx;
    fmBuf[(size_t)gq * NSV + s] = ((s < cnt) ? 1.f : 0.f) + inv;
  }
  fmSum[gq] = (float)cnt + (float)NSV * inv;
}

// ---------------- K2: gather + rel-weighted pooling (wave32 per query) -------
__global__ __launch_bounds__(256) void pp_gather_pool(
    const float* __restrict__ qxyz, const float* __restrict__ sxyz,
    const float* __restrict__ feat, const int* __restrict__ idxBuf,
    const float* __restrict__ fmBuf, float* __restrict__ pooled,
    float* __restrict__ gse) {
  __shared__ float gseL[CCV];
  for (int i = threadIdx.x; i < CCV; i += 256) gseL[i] = 0.f;
  __syncthreads();

  const int wave = threadIdx.x >> 5;
  const int lane = threadIdx.x & 31;
  const int gq = blockIdx.x * 8 + wave;      // 8 queries per block (same batch)
  const int bB = (blockIdx.x * 8) / N1V;     // uniform batch id for this block
  const int q  = gq % N1V;

  // lane s owns sample s of this wave's query
  const int   myIdx = idxBuf[(size_t)gq * NSV + lane];
  const float myFm  = fmBuf[(size_t)gq * NSV + lane];
  const float qx = qxyz[(size_t)gq * 3 + 0];
  const float qy = qxyz[(size_t)gq * 3 + 1];
  const float qz = qxyz[(size_t)gq * 3 + 2];
  const size_t sb = ((size_t)bB * N2V + myIdx) * 3;
  const float invR = 1.f / RADIUSV;
  const float rx = (sxyz[sb + 0] - qx) * invR;
  const float ry = (sxyz[sb + 1] - qy) * invR;
  const float rz = (sxyz[sb + 2] - qz) * invR;

  const float* fb = feat + (size_t)bB * CCV * N2V;
  float accA[9], accF[9];
#pragma unroll
  for (int j = 0; j < 9; ++j) { accA[j] = 0.f; accF[j] = 0.f; }

  for (int s = 0; s < NSV; ++s) {
    const int   i   = __shfl(myIdx, s, 32);
    const float fms = __shfl(myFm, s, 32);
    const float r0  = __shfl(rx, s, 32);
    const float r1  = __shfl(ry, s, 32);
    const float r2v = __shfl(rz, s, 32);
    const int iNext = __shfl(myIdx, (s + 1) & 31, 32);
    __builtin_prefetch(&fb[(size_t)lane * N2V + iNext], 0, 0); // global_prefetch_b8
#pragma unroll
    for (int j = 0; j < 9; ++j) {
      const int c = lane + 32 * j;
      const float val = fb[(size_t)c * N2V + i];  // L2-resident scattered gather
      const int k = c % 3;
      const float rel = (k == 0) ? r0 : ((k == 1) ? r1 : r2v);
      const float a = val * rel;
      accA[j] += a;                // unmasked -> squeeze
      accF[j] += a * fms;          // fm-masked -> output numerator
    }
  }
#pragma unroll
  for (int j = 0; j < 9; ++j) {
    const int c = lane + 32 * j;
    pooled[((size_t)bB * CCV + c) * N1V + q] = accF[j];
    atomicAdd(&gseL[c], accA[j]);               // ds_add_f32
  }
  __syncthreads();
  for (int i = threadIdx.x; i < CCV; i += 256)
    atomicAdd(&gse[bB * CCV + i], gseL[i]);     // global_atomic_add_f32
}

// ---------------- K3: squeeze-excite MLP via v_wmma_f32_16x16x32_f16 --------
// All operands staged once into zero-padded LDS tiles -> the WMMA loops are
// pure unconditional ds_load + v_wmma (no per-element exec masking).
// M = batch (4, padded to 16); GEMM1: K=288 (9 steps) x N=18 (2 tiles);
// GEMM2: K=18 (padded 32)     x N=288 (18 tiles). Single wave32.
__global__ __launch_bounds__(32) void pp_se_wmma(
    const float* __restrict__ gse, const float* __restrict__ w1,
    const float* __restrict__ w2, float* __restrict__ excite) {
  __shared__ _Float16 aP[16][CCV];    // gse padded to 16 rows        (9 KB)
  __shared__ _Float16 w1P[32][CCV];   // w1 padded 18 -> 32 rows     (18 KB)
  __shared__ _Float16 w2P[CCV][32];   // w2 padded 18 -> 32 cols     (18 KB)
  __shared__ float    hL[16][32];     // hidden after ReLU            (2 KB)
  const int lane = threadIdx.x & 31;
  const int m = lane & 15;
  const bool hi = lane >= 16;
  const float invPool = 1.f / ((float)N1V * (float)NSV);

  // ---- stage with zero padding (all regions disjoint; one barrier) ----
  for (int i = lane; i < BB * CCV; i += 32)
    aP[i / CCV][i % CCV] = (_Float16)(gse[i] * invPool);
  for (int i = lane; i < (16 - BB) * CCV; i += 32)
    aP[BB + i / CCV][i % CCV] = (_Float16)0.f;
  for (int i = lane; i < 18 * CCV; i += 32)
    w1P[i / CCV][i % CCV] = (_Float16)w1[i];
  for (int i = lane; i < (32 - 18) * CCV; i += 32)
    w1P[18 + i / CCV][i % CCV] = (_Float16)0.f;
  for (int i = lane; i < CCV * 18; i += 32)
    w2P[i / 18][i % 18] = (_Float16)w2[i];
  for (int i = lane; i < CCV * 14; i += 32)
    w2P[i / 14][18 + i % 14] = (_Float16)0.f;
  __syncthreads();

  // ---- GEMM1: h = relu(gse @ w1^T) ----
  v8f acc0 = {}; v8f acc1 = {};
  for (int t = 0; t < CCV / 32; ++t) {          // 9 K-steps of 32
    v16h a, b0, b1;
#pragma unroll
    for (int i = 0; i < 16; ++i) {              // ISA A-layout: K striped in-lane
      const int klA = (i < 8) ? ((hi ? 8 : 0) + i) : (16 + (hi ? 8 : 0) + (i - 8));
      a[i] = aP[m][t * 32 + klA];
      const int klB = (i < 8) ? ((hi ? 8 : 0) + i) : (16 + (hi ? 8 : 0) + (i - 8));        // ISA B-layout: N across lanes
      b0[i] = w1P[m][t * 32 + klB];
      b1[i] = w1P[16 + m][t * 32 + klB];
    }
    acc0 = __builtin_amdgcn_wmma_f32_16x16x32_f16(false, a, false, b0, (short)0, acc0, false, false);
    acc1 = __builtin_amdgcn_wmma_f32_16x16x32_f16(false, a, false, b1, (short)0, acc1, false, false);
  }
#pragma unroll
  for (int r = 0; r < 8; ++r) {                 // D-layout: reg r -> M = r + 8*hi
    const int M = r + (hi ? 8 : 0);
    hL[M][m]      = fmaxf(acc0[r], 0.f);
    hL[M][16 + m] = fmaxf(acc1[r], 0.f);        // cols 18..31 are exactly 0
  }
  __syncthreads();

  // ---- GEMM2: e = sigmoid(h @ w2^T) ----
  v16h a2;
#pragma unroll
  for (int i = 0; i < 16; ++i) {
    const int klA = (i < 8) ? ((hi ? 8 : 0) + i) : (16 + (hi ? 8 : 0) + (i - 8));
    a2[i] = (_Float16)hL[m][klA];
  }
  for (int nt = 0; nt < CCV / 16; ++nt) {       // 18 N-tiles over C
    const int c = nt * 16 + m;
    v16h b2;
#pragma unroll
    for (int i = 0; i < 16; ++i) {
      const int j = (i < 8) ? ((hi ? 8 : 0) + i) : (16 + (hi ? 8 : 0) + (i - 8));
      b2[i] = w2P[c][j];
    }
    v8f acc = {};
    acc = __builtin_amdgcn_wmma_f32_16x16x32_f16(false, a2, false, b2, (short)0, acc, false, false);
#pragma unroll
    for (int r = 0; r < 8; ++r) {
      const int M = r + (hi ? 8 : 0);
      if (M < BB) excite[M * CCV + c] = 1.f / (1.f + __expf(-acc[r]));
    }
  }
}

// ---------------- K4: apply excite + fm normalization; channel stats --------
__global__ __launch_bounds__(256) void pp_apply_stats(
    float* __restrict__ pooled, const float* __restrict__ excite,
    const float* __restrict__ fmSum, float* __restrict__ chSum,
    float* __restrict__ chSumSq) {
  __shared__ float sS[256], sQ[256];
  const size_t t = (size_t)blockIdx.x * 256 + threadIdx.x;
  const int q = (int)(t % N1V);
  const int c = (int)((t / N1V) % CCV);
  const int b = (int)(t / ((size_t)N1V * CCV));
  const float v = pooled[t] * excite[b * CCV + c] / fmSum[b * N1V + q];
  pooled[t] = v;                                 // in place, deterministic
  sS[threadIdx.x] = v;
  sQ[threadIdx.x] = v * v;
  __syncthreads();
  for (int o = 128; o > 0; o >>= 1) {
    if (threadIdx.x < o) {
      sS[threadIdx.x] += sS[threadIdx.x + o];
      sQ[threadIdx.x] += sQ[threadIdx.x + o];
    }
    __syncthreads();
  }
  if (threadIdx.x == 0) {                        // whole block shares one channel
    atomicAdd(&chSum[c], sS[0]);
    atomicAdd(&chSumSq[c], sQ[0]);
  }
}

// ---------------- K5: per-channel normalize + affine + ReLU -----------------
__global__ __launch_bounds__(256) void pp_norm_out(
    const float* __restrict__ pooled, const float* __restrict__ chSum,
    const float* __restrict__ chSumSq, const float* __restrict__ gamma,
    const float* __restrict__ beta, float* __restrict__ out) {
  const size_t t = (size_t)blockIdx.x * 256 + threadIdx.x;
  const int c = (int)((t / N1V) % CCV);
  const float n = (float)(BB * N1V);
  const float mean = chSum[c] / n;
  const float var = chSumSq[c] / n - mean * mean;
  const float v = gamma[c] * (pooled[t] - mean) * rsqrtf(var + EPSV) + beta[c];
  out[t] = fmaxf(v, 0.f);
}

// ---------------- launch -----------------------------------------------------
extern "C" void kernel_launch(void* const* d_in, const int* in_sizes, int n_in,
                              void* d_out, int out_size, void* d_ws, size_t ws_size,
                              hipStream_t stream) {
  const float* qxyz  = (const float*)d_in[0];
  const float* sxyz  = (const float*)d_in[1];
  const float* qmask = (const float*)d_in[2];
  const float* smask = (const float*)d_in[3];
  const float* feat  = (const float*)d_in[4];
  const float* w1    = (const float*)d_in[5];
  const float* w2    = (const float*)d_in[6];
  const float* gamma = (const float*)d_in[7];
  const float* beta  = (const float*)d_in[8];
  float* out = (float*)d_out;

  char* ws = (char*)d_ws;
  size_t oIdx = 0;
  size_t oFm  = oIdx + (size_t)BB * N1V * NSV * sizeof(int);
  size_t oFmS = oFm  + (size_t)BB * N1V * NSV * sizeof(float);
  size_t oPo  = oFmS + (size_t)BB * N1V * sizeof(float);
  size_t oGse = oPo  + (size_t)BB * CCV * N1V * sizeof(float);
  size_t oExc = oGse + (size_t)BB * CCV * sizeof(float);
  size_t oChS = oExc + (size_t)BB * CCV * sizeof(float);
  size_t oChQ = oChS + (size_t)CCV * sizeof(float);

  int*   idxBuf  = (int*)(ws + oIdx);
  float* fmBuf   = (float*)(ws + oFm);
  float* fmSum   = (float*)(ws + oFmS);
  float* pooled  = (float*)(ws + oPo);
  float* gse     = (float*)(ws + oGse);
  float* excite  = (float*)(ws + oExc);
  float* chSum   = (float*)(ws + oChS);
  float* chSumSq = (float*)(ws + oChQ);

  pp_init<<<(BB * CCV + 255) / 256, 256, 0, stream>>>(gse, chSum, chSumSq);
  pp_ballquery<<<(BB * N1V) / 256, 256, 0, stream>>>(qxyz, sxyz, qmask, smask,
                                                     idxBuf, fmBuf, fmSum);
  pp_gather_pool<<<(BB * N1V) / 8, 256, 0, stream>>>(qxyz, sxyz, feat, idxBuf,
                                                     fmBuf, pooled, gse);
  pp_se_wmma<<<1, 32, 0, stream>>>(gse, w1, w2, excite);
  const size_t nel = (size_t)BB * CCV * N1V;
  pp_apply_stats<<<(unsigned)(nel / 256), 256, 0, stream>>>(pooled, excite, fmSum,
                                                            chSum, chSumSq);
  pp_norm_out<<<(unsigned)(nel / 256), 256, 0, stream>>>(pooled, chSum, chSumSq,
                                                         gamma, beta, out);
}
